// BiLSTMCRF_20109036880076
// MI455X (gfx1250) — matrix operand order repaired and checked by reading, not compiled, not run on hardware
//
#include <hip/hip_runtime.h>
#include <math.h>

// ---------------------------------------------------------------------------
// BiLSTM-CRF forward (NLL mean) for MI455X / gfx1250, wave32, bf16 WMMA.
//   V=50000 E=300(pad 320) H=128 4H=512 K=20(pad 32) B=128 T=512 BT=65536
// Pipeline (all on `stream`):
//   1. convert embedding table + all weights to bf16 (K-padded), sum biases
//   2. gin_l0{f,r} = gather-GEMM(table[x]) via v_wmma_f32_16x16x32_bf16
//   3. LSTM recurrence layer0 (grid 8x2): Whh resident in 320KB LDS,
//      per-step gin tiles streamed by the Tensor Data Mover (TENSORcnt),
//      double-buffered so DMA overlaps the WMMA + gate math.
//   4. gin_l1{f,r} = GEMM(h0)      5. recurrence layer1
//   6. emissions = GEMM(h1, W2t)   7. CRF per-batch NLL  8. mean -> d_out[0]
// Workspace assumption: ~336 MB (two fp32 gin buffers are the bulk).
// ---------------------------------------------------------------------------

typedef __bf16 v16bf __attribute__((ext_vector_type(16)));
typedef __bf16 v8bf  __attribute__((ext_vector_type(8)));
typedef float  v8f   __attribute__((ext_vector_type(8)));
typedef unsigned int u32x4 __attribute__((ext_vector_type(4)));
typedef int          i32x8 __attribute__((ext_vector_type(8)));
typedef int          i32x4 __attribute__((ext_vector_type(4)));

#define Bc   128
#define Tc   512
#define BTc  65536
#define Hc   128
#define G4c  512
#define H2c  256
#define Ec   300
#define EPc  320
#define Vc   50000
#define Kc   20
#define KPc  32

// ---------------- conversion / small utility kernels -----------------------

__global__ __launch_bounds__(256) void cvt_pad_bf16_k(
    const float* __restrict__ src, __bf16* __restrict__ dst,
    int src_rows, int src_cols, int dst_rows, int dst_cols)
{
    long long i = (long long)blockIdx.x * 256 + threadIdx.x;
    long long n = (long long)dst_rows * dst_cols;
    if (i >= n) return;
    int r = (int)(i / dst_cols), c = (int)(i % dst_cols);
    float v = (r < src_rows && c < src_cols) ? src[(long long)r * src_cols + c] : 0.f;
    dst[i] = (__bf16)v;
}

__global__ __launch_bounds__(256) void bias_sum_k(
    const float* __restrict__ a, const float* __restrict__ b,
    float* __restrict__ o, int n)
{
    int i = blockIdx.x * 256 + threadIdx.x;
    if (i < n) o[i] = a[i] + b[i];
}

__global__ __launch_bounds__(256) void pad_f32_k(
    const float* __restrict__ src, float* __restrict__ dst, int n_src, int n_dst)
{
    int i = blockIdx.x * 256 + threadIdx.x;
    if (i < n_dst) dst[i] = (i < n_src) ? src[i] : 0.f;
}

// ---------------- generic bf16 WMMA GEMM:  C = A(rows) x W^T + bias --------
// A: bf16 row-major [*, lda] (optionally gathered by gidx), W: bf16 [N][Kd],
// C: f32 [M][ldc].  One 16x16 tile per wave, 8 waves/block.

__global__ __launch_bounds__(256) void gemm_bf16_wmma_k(
    const __bf16* __restrict__ A, int lda, const int* __restrict__ gidx,
    const __bf16* __restrict__ W, int Kd,
    const float* __restrict__ bias,
    float* __restrict__ C, int ldc,
    int Mtiles, int Ntiles)
{
    int wv = threadIdx.x >> 5, lane = threadIdx.x & 31;
    long long tile = (long long)blockIdx.x * 8 + wv;
    if (tile >= (long long)Mtiles * Ntiles) return;   // wave-uniform exit
    int mt = (int)(tile / Ntiles), nt = (int)(tile % Ntiles);
    int lr = lane & 15, hi = lane >> 4;
    int m  = mt * 16 + lr;
    long long arow = gidx ? (long long)gidx[m] : (long long)m;
    const __bf16* Ap = A + arow * lda;
    const __bf16* Wp = W + (long long)(nt * 16 + lr) * Kd + hi * 16;
    int khalf = hi * 8;

    v8f acc = {};
    for (int k0 = 0; k0 < Kd; k0 += 32) {
        // A fragment: lane holds row m, 16 contiguous K values [hi*16..+15]
        v8bf alo = *(const v8bf*)(Ap + k0 + hi * 16);
        v8bf ahi = *(const v8bf*)(Ap + k0 + hi * 16 + 8);
        v16bf a = __builtin_shufflevector(alo, ahi,
                     0,1,2,3,4,5,6,7,8,9,10,11,12,13,14,15);
        // B fragment: lane holds column n, 16 contiguous K values
        v16bf b = *(const v16bf*)(Wp + k0);
        acc = __builtin_amdgcn_wmma_f32_16x16x32_bf16(
                  false, a, false, b, (short)0, acc, false, false);
    }
    int n = nt * 16 + lr;
    float bv = bias ? bias[n] : 0.f;
    long long mbase = (long long)mt * 16 + hi * 8;
    #pragma unroll
    for (int r = 0; r < 8; ++r)
        C[(mbase + r) * ldc + n] = acc[r] + bv;
}

// ---------------- LSTM recurrence ------------------------------------------
// grid (8, 2): blockIdx.x = 16-row batch slice, blockIdx.y = direction.
// LDS: Whh [512][128] bf16 (128KB) + h [16][128] bf16 + c [16][128] f32
//      + gate staging [16][512] f32 + 2x TDM gin tile [16][512] f32
//      = 236KB (CDNA5 WGP has 320KB).
// Per-step gin tile (16 rows x 512 f32, row stride T*512) is DMA'd by the
// Tensor Data Mover (issued by wave 0, tracked with TENSORcnt) one step
// ahead into the alternate buffer.

__device__ __forceinline__ float sigm(float x) { return 1.f / (1.f + __expf(-x)); }

__global__ __launch_bounds__(512) void lstm_rec_k(
    const float* __restrict__ ginF, const float* __restrict__ ginR,
    const __bf16* __restrict__ whhF, const __bf16* __restrict__ whhR,
    __bf16* __restrict__ hout, int T)
{
    const int dir = blockIdx.y;
    const float*  gin = dir ? ginR : ginF;
    const __bf16* whh = dir ? whhR : whhF;
    const int bblk = blockIdx.x;

    extern __shared__ char smem[];
    __bf16* whh_s = (__bf16*)smem;                          // 512*128 bf16
    __bf16* h_s   = (__bf16*)(smem + 131072);               // 16*128 bf16
    float*  c_s   = (float*)(smem + 131072 + 4096);         // 16*128 f32
    float*  g_s   = (float*)(smem + 131072 + 4096 + 8192);  // 16*512 f32
    float*  gbuf  = (float*)(smem + 131072 + 4096 + 8192 + 32768); // 2x 16*512 f32

    int tid = threadIdx.x;

    // ---- TDM descriptor issue: 2D tile, 512 f32 (dim0) x 16 rows (dim1) ----
    auto tdm_issue = [&](int tt, int buf) {
        const float* gsrc = gin + ((long long)(bblk * 16) * T + tt) * (long long)G4c;
        unsigned long long ga = (unsigned long long)(size_t)gsrc;
        unsigned lds = (unsigned)(size_t)(gbuf + buf * (16 * G4c));
        const long long stride0 = (long long)T * G4c;       // elements between rows
        u32x4 g0;
        g0[0] = 1u;                                          // count=1, load, no gather
        g0[1] = lds;                                         // lds_addr (bytes)
        g0[2] = (unsigned)ga;                                // global_addr[31:0]
        g0[3] = (unsigned)((ga >> 32) & 0x01FFFFFFull) | 0x80000000u; // [56:32] | type=2
        i32x8 g1;
        g1[0] = 0x00020000;                                  // data_size=4B
        g1[1] = (int)((G4c & 0xFFFF) << 16);                 // tensor_dim0[15:0]
        g1[2] = (int)(((unsigned)G4c >> 16) |
                      (((unsigned)BTc & 0xFFFFu) << 16));    // td0[31:16] | td1[15:0]
        g1[3] = (int)((((unsigned)BTc >> 16) & 0xFFFFu) |
                      (((unsigned)G4c & 0xFFFFu) << 16));    // td1[31:16] | tile_dim0
        g1[4] = 16;                                          // tile_dim1=16, tile_dim2=0
        g1[5] = (int)(unsigned)(stride0 & 0xFFFFFFFFll);     // tensor_dim0_stride[31:0]
        g1[6] = (int)(unsigned)((stride0 >> 32) & 0xFFFFll); // stride[47:32]
        g1[7] = 0;
        i32x4 gz4 = {0, 0, 0, 0};                            // 2D: groups 2/3 unused
        i32x8 gz8 = {0, 0, 0, 0, 0, 0, 0, 0};                // (count=1: ignored)
        __builtin_amdgcn_tensor_load_to_lds(g0, g1, gz4, gz4, gz8, 0);
    };

    // prefetch gin tile for step 0 while Whh is staged
    if (tid < 32) tdm_issue(dir ? T - 1 : 0, 0);

    { // stage Whh into LDS (16B vector copies)
        const v8bf* src = (const v8bf*)whh;
        v8bf* dst = (v8bf*)whh_s;
        for (int i = tid; i < (G4c * Hc) / 8; i += 512) dst[i] = src[i];
    }
    for (int i = tid; i < 16 * Hc; i += 512) { h_s[i] = (__bf16)0.f; c_s[i] = 0.f; }
    __syncthreads();

    int wv = tid >> 5, lane = tid & 31;
    int lr = lane & 15, hi = lane >> 4;
    int khalf = hi * 8;

    for (int s = 0; s < T; ++s) {
        int t = dir ? (T - 1 - s) : s;
        // wave 0 waits for this step's DMA, barrier releases everyone,
        // then next step's tile is DMA'd into the alternate buffer.
        if (tid < 32) __builtin_amdgcn_s_wait_tensorcnt(0);
        __syncthreads();
        if (tid < 32 && s + 1 < T) tdm_issue(dir ? t - 1 : t + 1, (s + 1) & 1);
        const float* gs = gbuf + (s & 1) * (16 * G4c);

        // A fragments of h (shared by this wave's two n-tiles)
        v16bf afrag[4];
        #pragma unroll
        for (int kc = 0; kc < 4; ++kc) {
            const __bf16* hp = h_s + lr * Hc + kc * 32;
            v8bf alo = *(const v8bf*)(hp + hi * 16);
            v8bf ahi = *(const v8bf*)(hp + hi * 16 + 8);
            afrag[kc] = __builtin_shufflevector(alo, ahi,
                           0,1,2,3,4,5,6,7,8,9,10,11,12,13,14,15);
        }
        #pragma unroll
        for (int it = 0; it < 2; ++it) {
            int nt = wv * 2 + it;
            v8f acc = {};
            #pragma unroll
            for (int kc = 0; kc < 4; ++kc) {
                v16bf b = *(const v16bf*)(whh_s + (nt * 16 + lr) * Hc + kc * 32 + hi * 16);
                acc = __builtin_amdgcn_wmma_f32_16x16x32_bf16(
                          false, afrag[kc], false, b, (short)0, acc, false, false);
            }
            int n = nt * 16 + lr;
            int mb = hi * 8;
            #pragma unroll
            for (int r = 0; r < 8; ++r)
                g_s[(mb + r) * G4c + n] = acc[r] + gs[(mb + r) * G4c + n];
        }
        __syncthreads();
        // elementwise gate update: 16*128 cells / 512 threads
        for (int i = tid; i < 16 * Hc; i += 512) {
            int m = i >> 7, j = i & 127;
            float gi = g_s[m * G4c + j];
            float gf = g_s[m * G4c + Hc + j];
            float gg = g_s[m * G4c + 2 * Hc + j];
            float go = g_s[m * G4c + 3 * Hc + j];
            float c  = sigm(gf) * c_s[i] + sigm(gi) * tanhf(gg);
            c_s[i] = c;
            float h = sigm(go) * tanhf(c);
            h_s[m * Hc + j] = (__bf16)h;
            hout[((long long)(bblk * 16 + m) * T + t) * H2c + dir * Hc + j] = (__bf16)h;
        }
        __syncthreads();
    }
}

// ---------------- CRF NLL per batch element (one wave32 per block) ---------

__global__ __launch_bounds__(32) void crf_k(
    const float* __restrict__ em,   // [B*T][32] padded emissions
    const int* __restrict__ tags,
    const float* __restrict__ startv, const float* __restrict__ endv,
    const float* __restrict__ trans, float* __restrict__ partial, int T)
{
    __shared__ float trans_s[Kc * Kc];
    __shared__ float alpha_s[Kc];
    __shared__ float red[32];
    __shared__ float s_score;
    int lane = threadIdx.x, b = blockIdx.x;
    const int*   tg  = tags + (long long)b * T;
    const float* emb = em + (long long)b * T * KPc;

    for (int i = lane; i < Kc * Kc; i += 32) trans_s[i] = trans[i];
    __syncthreads();

    // gold path score
    float sc = 0.f;
    for (int t = lane; t < T; t += 32) {
        sc += emb[(long long)t * KPc + tg[t]];
        if (t + 1 < T) sc += trans_s[tg[t] * Kc + tg[t + 1]];
    }
    red[lane] = sc; __syncthreads();
    for (int st = 16; st > 0; st >>= 1) { if (lane < st) red[lane] += red[lane + st]; __syncthreads(); }
    if (lane == 0) s_score = red[0] + startv[tg[0]] + endv[tg[T - 1]];

    // forward algorithm (lane j < 20 owns alpha[j])
    if (lane < Kc) alpha_s[lane] = startv[lane] + emb[lane];
    __syncthreads();
    for (int t = 1; t < T; ++t) {
        float nv = 0.f;
        if (lane < Kc) {
            float mx = -3.4e38f;
            #pragma unroll
            for (int i = 0; i < Kc; ++i) mx = fmaxf(mx, alpha_s[i] + trans_s[i * Kc + lane]);
            float sm = 0.f;
            #pragma unroll
            for (int i = 0; i < Kc; ++i) sm += __expf(alpha_s[i] + trans_s[i * Kc + lane] - mx);
            nv = mx + __logf(sm) + emb[(long long)t * KPc + lane];
        }
        __syncthreads();
        if (lane < Kc) alpha_s[lane] = nv;
        __syncthreads();
    }
    float v = (lane < Kc) ? alpha_s[lane] + endv[lane] : -3.4e38f;
    red[lane] = v; __syncthreads();
    for (int st = 16; st > 0; st >>= 1) { if (lane < st) red[lane] = fmaxf(red[lane], red[lane + st]); __syncthreads(); }
    float mx = red[0]; __syncthreads();
    red[lane] = (lane < Kc) ? __expf(v - mx) : 0.f; __syncthreads();
    for (int st = 16; st > 0; st >>= 1) { if (lane < st) red[lane] += red[lane + st]; __syncthreads(); }
    if (lane == 0) partial[b] = (mx + __logf(red[0])) - s_score;
}

__global__ __launch_bounds__(128) void mean128_k(const float* __restrict__ p,
                                                 float* __restrict__ out)
{
    __shared__ float s[128];
    int t = threadIdx.x;
    s[t] = p[t]; __syncthreads();
    for (int st = 64; st > 0; st >>= 1) { if (t < st) s[t] += s[t + st]; __syncthreads(); }
    if (t == 0) out[0] = s[0] * (1.f / 128.f);
}

// ---------------------------------------------------------------------------

static inline size_t align256(size_t x) { return (x + 255) & ~(size_t)255; }

extern "C" void kernel_launch(void* const* d_in, const int* in_sizes, int n_in,
                              void* d_out, int out_size, void* d_ws, size_t ws_size,
                              hipStream_t stream) {
    (void)in_sizes; (void)n_in; (void)out_size; (void)ws_size;
    const int*   x      = (const int*)  d_in[0];
    // d_in[1] = lengths (unused; sequences are full length)
    const int*   tags   = (const int*)  d_in[2];
    const float* emb    = (const float*)d_in[3];
    const float* h2t_w  = (const float*)d_in[4];
    const float* h2t_b  = (const float*)d_in[5];
    const float* startv = (const float*)d_in[6];
    const float* endv   = (const float*)d_in[7];
    const float* trans  = (const float*)d_in[8];
    const float* wih[4] = { (const float*)d_in[9],  (const float*)d_in[13],
                            (const float*)d_in[17], (const float*)d_in[21] };
    const float* whh[4] = { (const float*)d_in[10], (const float*)d_in[14],
                            (const float*)d_in[18], (const float*)d_in[22] };
    const float* bih[4] = { (const float*)d_in[11], (const float*)d_in[15],
                            (const float*)d_in[19], (const float*)d_in[23] };
    const float* bhh[4] = { (const float*)d_in[12], (const float*)d_in[16],
                            (const float*)d_in[20], (const float*)d_in[24] };

    uint8_t* ws = (uint8_t*)d_ws;
    size_t off = 0;
    auto take = [&](size_t bytes) { size_t o = off; off = align256(off + bytes); return o; };

    __bf16* tbl    = (__bf16*)(ws + take((size_t)Vc * EPc * 2));     // 32 MB
    __bf16* wih0f  = (__bf16*)(ws + take((size_t)G4c * EPc * 2));
    __bf16* wih0r  = (__bf16*)(ws + take((size_t)G4c * EPc * 2));
    __bf16* whhB[4];
    for (int i = 0; i < 4; ++i) whhB[i] = (__bf16*)(ws + take((size_t)G4c * Hc * 2));
    __bf16* wih1f  = (__bf16*)(ws + take((size_t)G4c * H2c * 2));
    __bf16* wih1r  = (__bf16*)(ws + take((size_t)G4c * H2c * 2));
    __bf16* w2t    = (__bf16*)(ws + take((size_t)KPc * H2c * 2));
    float*  biasB[4];
    for (int i = 0; i < 4; ++i) biasB[i] = (float*)(ws + take((size_t)G4c * 4));
    float*  bias2t = (float*)(ws + take((size_t)KPc * 4));
    float*  ginF   = (float*)(ws + take((size_t)BTc * G4c * 4));     // 134 MB
    float*  ginR   = (float*)(ws + take((size_t)BTc * G4c * 4));     // 134 MB
    __bf16* h0     = (__bf16*)(ws + take((size_t)BTc * H2c * 2));    //  33 MB
    __bf16* h1     = h0;                       // reuse: h0 dead after gin_l1
    float*  emis   = ginF;                     // reuse gin_f after l1 rec
    float*  partial= ginR;                     // reuse gin_r after CRF input

    // ---- 1. precision conversion ----
    auto cvtN = [](long long n) { return (unsigned)((n + 255) / 256); };
    cvt_pad_bf16_k<<<cvtN((long long)Vc * EPc), 256, 0, stream>>>(emb, tbl, Vc, Ec, Vc, EPc);
    cvt_pad_bf16_k<<<cvtN((long long)G4c * EPc), 256, 0, stream>>>(wih[0], wih0f, G4c, Ec, G4c, EPc);
    cvt_pad_bf16_k<<<cvtN((long long)G4c * EPc), 256, 0, stream>>>(wih[1], wih0r, G4c, Ec, G4c, EPc);
    for (int i = 0; i < 4; ++i)
        cvt_pad_bf16_k<<<cvtN((long long)G4c * Hc), 256, 0, stream>>>(whh[i], whhB[i], G4c, Hc, G4c, Hc);
    cvt_pad_bf16_k<<<cvtN((long long)G4c * H2c), 256, 0, stream>>>(wih[2], wih1f, G4c, H2c, G4c, H2c);
    cvt_pad_bf16_k<<<cvtN((long long)G4c * H2c), 256, 0, stream>>>(wih[3], wih1r, G4c, H2c, G4c, H2c);
    cvt_pad_bf16_k<<<cvtN((long long)KPc * H2c), 256, 0, stream>>>(h2t_w, w2t, Kc, H2c, KPc, H2c);
    for (int i = 0; i < 4; ++i)
        bias_sum_k<<<2, 256, 0, stream>>>(bih[i], bhh[i], biasB[i], G4c);
    pad_f32_k<<<1, 256, 0, stream>>>(h2t_b, bias2t, Kc, KPc);

    const int MT = BTc / 16;                  // 4096 M-tiles
    const unsigned smemRec = 131072 + 4096 + 8192 + 32768 + 65536;  // 236 KB

    // ---- 2. layer-0 input projection (fused embedding gather) ----
    {
        int NT = G4c / 16;                    // 32
        unsigned blks = (unsigned)(((long long)MT * NT + 7) / 8);
        gemm_bf16_wmma_k<<<blks, 256, 0, stream>>>(tbl, EPc, x, wih0f, EPc, biasB[0], ginF, G4c, MT, NT);
        gemm_bf16_wmma_k<<<blks, 256, 0, stream>>>(tbl, EPc, x, wih0r, EPc, biasB[1], ginR, G4c, MT, NT);
    }
    // ---- 3. layer-0 recurrence (both directions concurrently) ----
    lstm_rec_k<<<dim3(Bc / 16, 2), 512, smemRec, stream>>>(ginF, ginR, whhB[0], whhB[1], h0, Tc);

    // ---- 4. layer-1 input projection ----
    {
        int NT = G4c / 16;
        unsigned blks = (unsigned)(((long long)MT * NT + 7) / 8);
        gemm_bf16_wmma_k<<<blks, 256, 0, stream>>>(h0, H2c, nullptr, wih1f, H2c, biasB[2], ginF, G4c, MT, NT);
        gemm_bf16_wmma_k<<<blks, 256, 0, stream>>>(h0, H2c, nullptr, wih1r, H2c, biasB[3], ginR, G4c, MT, NT);
    }
    // ---- 5. layer-1 recurrence ----
    lstm_rec_k<<<dim3(Bc / 16, 2), 512, smemRec, stream>>>(ginF, ginR, whhB[2], whhB[3], h1, Tc);

    // ---- 6. emissions ----
    {
        int NT = KPc / 16;                    // 2
        unsigned blks = (unsigned)(((long long)MT * NT + 7) / 8);
        gemm_bf16_wmma_k<<<blks, 256, 0, stream>>>(h1, H2c, nullptr, w2t, H2c, bias2t, emis, KPc, MT, NT);
    }
    // ---- 7/8. CRF NLL + mean ----
    crf_k<<<Bc, 32, 0, stream>>>(emis, tags, startv, endv, trans, partial, Tc);
    mean128_k<<<1, 128, 0, stream>>>(partial, (float*)d_out);
}
